// MultiHeadAttention_57320633532459
// MI455X (gfx1250) — matrix operand order repaired and checked by reading, not compiled, not run on hardware
//
#include <hip/hip_runtime.h>
#include <cstdint>

// ---------------------------------------------------------------- types
typedef __attribute__((ext_vector_type(16))) __bf16 v16bf;
typedef __attribute__((ext_vector_type(8)))  float  v8f;

union BF16x16 { v16bf v; uint4 q[2]; unsigned short h[16]; };
union F8      { v8f   v; float f[8]; };

__device__ __forceinline__ unsigned short f2bf_bits(float x) {
  union { float f; unsigned u; } a; a.f = x;
  unsigned r = a.u + 0x7FFFu + ((a.u >> 16) & 1u);   // round-to-nearest-even
  return (unsigned short)(r >> 16);
}

__device__ __forceinline__ v8f wmma_bf16(v16bf a, v16bf b, v8f c) {
  // D = A(16x32 bf16) * B(32x16 bf16) + C(16x16 f32)
  return __builtin_amdgcn_wmma_f32_16x16x32_bf16(false, a, false, b, (short)0, c, false, false);
}

// ---------------------------------------------------------------- async LDS staging
#if __has_builtin(__builtin_amdgcn_global_load_async_to_lds_b128)
#define ATHENA_ASYNC_LDS 1
typedef int v4i_t __attribute__((ext_vector_type(4)));
typedef __attribute__((address_space(1))) v4i_t* gptr_v4i;   // global (AS1) int4*
typedef __attribute__((address_space(3))) v4i_t* lptr_v4i;   // LDS (AS3) int4*
#endif

// copy 32 bytes (16 bf16) global -> LDS, per-thread addresses
__device__ __forceinline__ void stage32B(unsigned short* dst, const unsigned short* src) {
#ifdef ATHENA_ASYNC_LDS
  __builtin_amdgcn_global_load_async_to_lds_b128((gptr_v4i)(src),     (lptr_v4i)(dst),     0, 0);
  __builtin_amdgcn_global_load_async_to_lds_b128((gptr_v4i)(src + 8), (lptr_v4i)(dst + 8), 0, 0);
#else
  uint4 a = ((const uint4*)src)[0];
  uint4 b = ((const uint4*)src)[1];
  ((uint4*)dst)[0] = a;
  ((uint4*)dst)[1] = b;
#endif
}

__device__ __forceinline__ void stage_wait() {
#ifdef ATHENA_ASYNC_LDS
#if __has_builtin(__builtin_amdgcn_s_wait_asynccnt)
  __builtin_amdgcn_s_wait_asynccnt(0);
#else
  asm volatile("s_wait_asynccnt 0x0" ::: "memory");
#endif
#endif
}

// A-matrix (16x32, bf16) fragment from row-major LDS [rows][stride]:
__device__ __forceinline__ v16bf lds_load_A(const unsigned short* base, int stride,
                                            int row0, int k0, int lane) {
  int m  = lane & 15;
  int kg = (lane & 16) >> 1;                 // 0 or 8
  const unsigned short* p = base + (row0 + m) * stride + k0 + kg;
  BF16x16 f;
  f.q[0] = *(const uint4*)(p);               // VGPR0-3 : K = k0+kg .. +7
  f.q[1] = *(const uint4*)(p + 16);          // VGPR4-7 : K = k0+16+kg .. +7
  return f.v;
}

// B-matrix (32x16, bf16) fragment from LDS holding B column-major: Bc[n][k]
__device__ __forceinline__ v16bf lds_load_B(const unsigned short* base, int stride,
                                            int n0, int k0, int lane) {
  int n  = lane & 15;
  int kk = k0 + ((lane & 16) >> 1);          // +0 or +8
  const unsigned short* p = base + (n0 + n) * stride + kk;
  BF16x16 f;
  f.q[0] = *(const uint4*)(p);               // K = kk .. kk+7
  f.q[1] = *(const uint4*)(p + 16);          // K = kk+16 .. kk+23
  return f.v;
}

// ---------------------------------------------------------------- constants
constexpr int Dm = 1024, NH = 16, DK = 64, Bb = 2, Ls = 2048;

// ---------------------------------------------------------------- fp32 -> bf16 (activations)
__global__ __launch_bounds__(256)
void cvt_f32_bf16_kernel(const float* __restrict__ in, unsigned short* __restrict__ out, int n) {
  int idx = (blockIdx.x * 256 + threadIdx.x) * 4;
  if (idx < n) {
    float4 f = *(const float4*)(in + idx);
    uint2 pk;
    pk.x = (unsigned)f2bf_bits(f.x) | ((unsigned)f2bf_bits(f.y) << 16);
    pk.y = (unsigned)f2bf_bits(f.z) | ((unsigned)f2bf_bits(f.w) << 16);
    *(uint2*)(out + idx) = pk;
  }
}

// ---------------------------------------------------------------- fp32 W[K][N] -> bf16 Wt[N][K]
// One-time weight transpose so GEMM B-fragments become contiguous LDS b128 loads.
__global__ __launch_bounds__(256)
void cvt_transpose_bf16_kernel(const float* __restrict__ W, unsigned short* __restrict__ Wt,
                               int K, int N) {
  __shared__ float tile[32][33];             // +1 pad: no bank conflicts on transpose
  const int k0 = blockIdx.y * 32, n0 = blockIdx.x * 32;
  const int t = threadIdx.x;
  {
    int r = t >> 3, c = (t & 7) * 4;         // 32x32 tile, 4 floats/thread
    float4 f = *(const float4*)(W + (size_t)(k0 + r) * N + n0 + c);
    tile[r][c + 0] = f.x; tile[r][c + 1] = f.y;
    tile[r][c + 2] = f.z; tile[r][c + 3] = f.w;
  }
  __syncthreads();
  {
    int n = t >> 3, c = (t & 7) * 4;
    uint2 pk;
    unsigned h0 = f2bf_bits(tile[c + 0][n]);
    unsigned h1 = f2bf_bits(tile[c + 1][n]);
    unsigned h2 = f2bf_bits(tile[c + 2][n]);
    unsigned h3 = f2bf_bits(tile[c + 3][n]);
    pk.x = h0 | (h1 << 16);
    pk.y = h2 | (h3 << 16);
    *(uint2*)(Wt + (size_t)(n0 + n) * K + k0 + c) = pk;
  }
}

// ---------------------------------------------------------------- GEMM 128x128, K-step 32
// C = X(MxK bf16 rm) @ W + bias, with W given PRE-TRANSPOSED as Wt[N][K].
// Double-buffered async LDS staging. Epilogue modes:
//  0: bf16 out at [b,h,l,dk]   (K,Q projections; scale applied)
//  1: bf16 out at [b,h,dk,l]   (V projection, pre-transposed for attention B-frags)
//  2: f32 out row-major MxN    (final output projection)
__global__ __launch_bounds__(256)
void gemm_bf16_wmma_kernel(const unsigned short* __restrict__ X,
                           const unsigned short* __restrict__ Wt,
                           const float* __restrict__ bias,
                           void* __restrict__ out,
                           int M, int N, int K, float scale, int mode) {
  __shared__ unsigned short Xs[2][128 * 32]; // A stage [m][k]
  __shared__ unsigned short Ws[2][128 * 32]; // B stage [n][k] (already column-major)

  const int tid = threadIdx.x, lane = tid & 31, wave = tid >> 5;
  const int wr = wave & 3, wc = wave >> 2;   // 4x2 wave grid -> 32x64 per wave
  const int n0blk = blockIdx.x * 128, m0blk = blockIdx.y * 128;

  const int rs = tid >> 1, cs = (tid & 1) * 16;            // staging coords (16 bf16/thread)
  const unsigned short* Xsrc = X  + (size_t)(m0blk + rs) * K + cs;
  const unsigned short* Wsrc = Wt + (size_t)(n0blk + rs) * K + cs;
  const int ldst = rs * 32 + cs;

  F8 acc[2][4];
#pragma unroll
  for (int i = 0; i < 2; ++i)
#pragma unroll
    for (int j = 0; j < 4; ++j)
#pragma unroll
      for (int v = 0; v < 8; ++v) acc[i][j].f[v] = 0.f;

  // prologue: stage k0 = 0 into buffer 0
  stage32B(&Xs[0][ldst], Xsrc);
  stage32B(&Ws[0][ldst], Wsrc);
  stage_wait();
  __syncthreads();

  int cur = 0;
  for (int k0 = 0; k0 < K; k0 += 32) {
    const int nxt = cur ^ 1;
    if (k0 + 32 < K) {                       // issue next tile while computing this one
      stage32B(&Xs[nxt][ldst], Xsrc + k0 + 32);
      stage32B(&Ws[nxt][ldst], Wsrc + k0 + 32);
    }
    v16bf a0 = lds_load_A(Xs[cur], 32, 32 * wr + 0, 0, lane);
    v16bf a1 = lds_load_A(Xs[cur], 32, 32 * wr + 16, 0, lane);
#pragma unroll
    for (int nt = 0; nt < 4; ++nt) {
      v16bf bfrag = lds_load_B(Ws[cur], 32, 64 * wc + 16 * nt, 0, lane);
      acc[0][nt].v = wmma_bf16(a0, bfrag, acc[0][nt].v);
      acc[1][nt].v = wmma_bf16(a1, bfrag, acc[1][nt].v);
    }
    stage_wait();                            // next tile landed in LDS
    __syncthreads();                         // everyone done reading cur
    cur = nxt;
  }

  // epilogue
  const int mb = ((lane & 16) >> 1);         // 0 or 8
  const int nn = lane & 15;
#pragma unroll
  for (int mt = 0; mt < 2; ++mt) {
#pragma unroll
    for (int nt = 0; nt < 4; ++nt) {
      int gcol = n0blk + 64 * wc + 16 * nt + nn;
      float bv = bias[gcol];
#pragma unroll
      for (int v = 0; v < 8; ++v) {
        int grow = m0blk + 32 * wr + 16 * mt + mb + v;
        float val = (acc[mt][nt].f[v] + bv) * scale;
        if (mode == 2) {
          ((float*)out)[(size_t)grow * N + gcol] = val;
        } else {
          int bidx = grow >> 11, l = grow & 2047;      // L = 2048
          int h = gcol >> 6, dk = gcol & 63;           // DK = 64
          unsigned short hv = f2bf_bits(val);
          if (mode == 0)
            ((unsigned short*)out)[(((size_t)(bidx * NH + h) * Ls + l) * DK + dk)] = hv;
          else
            ((unsigned short*)out)[(((size_t)(bidx * NH + h) * DK + dk) * Ls + l)] = hv;
        }
      }
    }
  }
}

// ---------------------------------------------------------------- flash attention
// Per block: one (b,h), 64 queries. Loop 64-key blocks with online softmax.
// Qp: [b,h,lp,dk] bf16 (pre-scaled by 1/sqrt(dk)); Kp: [b,h,l,dk]; Vp: [b,h,dk,l]
__global__ __launch_bounds__(256)
void attn_flash_kernel(const unsigned short* __restrict__ Qp,
                       const unsigned short* __restrict__ Kp,
                       const unsigned short* __restrict__ Vp,
                       const float* __restrict__ mask_attn,
                       const float* __restrict__ mask_pad,
                       unsigned short* __restrict__ Obuf) {
  __shared__ unsigned short Qs[64 * 64];     // [p][d]
  __shared__ unsigned short Ks[64 * 64];     // [l][d] == column-major of B for S
  __shared__ unsigned short Vs[64 * 64];     // [d][l] == column-major of B for O
  __shared__ float          Ss[64 * 64];     // scores f32 [p][l]
  __shared__ unsigned short Al[64 * 64];     // alpha bf16 [p][l]
  __shared__ float          rowScale[64];

  const int tid = threadIdx.x, lane = tid & 31, wave = tid >> 5;
  const int wr = wave & 3, wc = wave >> 2;   // wave: 16 q-rows x 32 cols
  const int q0 = blockIdx.x * 64, h = blockIdx.y, b = blockIdx.z;

  const unsigned short* Qh = Qp + (((size_t)(b * NH + h) * Ls + q0) * DK);
  const unsigned short* Kh = Kp + ((size_t)(b * NH + h) * Ls * DK);
  const unsigned short* Vh = Vp + ((size_t)(b * NH + h) * DK * Ls);

  const int rstg = tid >> 2, cstg = (tid & 3) * 16;        // staging coords
  const int ldsoff = rstg * 64 + cstg;

  // stage Q tile once (async)
  stage32B(&Qs[ldsoff], Qh + ldsoff);
  stage_wait();
  __syncthreads();
  const v16bf qa0 = lds_load_A(Qs, 64, 16 * wr, 0, lane);
  const v16bf qa1 = lds_load_A(Qs, 64, 16 * wr, 32, lane);

  F8 oacc[2];
#pragma unroll
  for (int ot = 0; ot < 2; ++ot)
#pragma unroll
    for (int v = 0; v < 8; ++v) oacc[ot].f[v] = 0.f;

  float m_i = -1e30f, l_i = 0.f;             // valid for tid < 64
  const v8f zacc = {0.f, 0.f, 0.f, 0.f, 0.f, 0.f, 0.f, 0.f};

  for (int kb = 0; kb < Ls / 64; ++kb) {
    __syncthreads();                         // readers of Ks/Vs done
    stage32B(&Ks[ldsoff], Kh + (size_t)(kb * 64 + rstg) * 64 + cstg);
    stage32B(&Vs[ldsoff], Vh + (size_t)rstg * Ls + kb * 64 + cstg);
    if (kb + 1 < Ls / 64 && tid < 64) {      // L2 prefetch next key/value block
      __builtin_prefetch(Kh + (size_t)((kb + 1) * 64) * 64 + tid * 64, 0, 1);
      __builtin_prefetch(Vh + (size_t)tid * Ls + (kb + 1) * 64, 0, 1);
    }
    stage_wait();
    __syncthreads();

    // S tiles: 16 q-rows x 2 l-tiles per wave
#pragma unroll
    for (int st = 0; st < 2; ++st) {
      int n0 = 32 * wc + 16 * st;
      v16bf b0 = lds_load_B(Ks, 64, n0, 0, lane);
      v16bf b1 = lds_load_B(Ks, 64, n0, 32, lane);
      F8 s;
      s.v = wmma_bf16(qa0, b0, zacc);
      s.v = wmma_bf16(qa1, b1, s.v);
      int mbase = 16 * wr + ((lane & 16) >> 1);
      int nc = n0 + (lane & 15);
#pragma unroll
      for (int v = 0; v < 8; ++v) Ss[(mbase + v) * 64 + nc] = s.f[v];
    }
    __syncthreads();

    // online-softmax row pass: thread p owns query row p
    if (tid < 64) {
      int p = tid;
      float* srow = Ss + p * 64;
      const float* ma = mask_attn + (size_t)(q0 + p) * Ls + kb * 64;
      const float* mp = mask_pad + (size_t)b * Ls + kb * 64;
      float bmax = -1e30f;
#pragma unroll 4
      for (int j = 0; j < 64; j += 4) {
        float4 s4 = *(float4*)(srow + j);
        float4 a4 = *(const float4*)(ma + j);
        float4 p4 = *(const float4*)(mp + j);
        s4.x += a4.x + p4.x; s4.y += a4.y + p4.y;
        s4.z += a4.z + p4.z; s4.w += a4.w + p4.w;
        bmax = fmaxf(bmax, fmaxf(fmaxf(s4.x, s4.y), fmaxf(s4.z, s4.w)));
        *(float4*)(srow + j) = s4;
      }
      float newm = fmaxf(m_i, bmax);
      float c = __expf(m_i - newm);
      float rsum = 0.f;
#pragma unroll 4
      for (int j = 0; j < 64; ++j) {
        float e = __expf(srow[j] - newm);
        Al[p * 64 + j] = f2bf_bits(e);
        rsum += e;
      }
      l_i = l_i * c + rsum;
      m_i = newm;
      rowScale[p] = c;
    }
    __syncthreads();

    // rescale O accumulators, then O += alpha @ V
    {
      int mbase = 16 * wr + ((lane & 16) >> 1);
      float cs[8];
#pragma unroll
      for (int v = 0; v < 8; ++v) cs[v] = rowScale[mbase + v];
#pragma unroll
      for (int ot = 0; ot < 2; ++ot)
#pragma unroll
        for (int v = 0; v < 8; ++v) oacc[ot].f[v] *= cs[v];

      v16bf a0 = lds_load_A(Al, 64, 16 * wr, 0, lane);
      v16bf a1 = lds_load_A(Al, 64, 16 * wr, 32, lane);
#pragma unroll
      for (int ot = 0; ot < 2; ++ot) {
        int n0 = 32 * wc + 16 * ot;
        v16bf b0 = lds_load_B(Vs, 64, n0, 0, lane);
        v16bf b1 = lds_load_B(Vs, 64, n0, 32, lane);
        oacc[ot].v = wmma_bf16(a0, b0, oacc[ot].v);
        oacc[ot].v = wmma_bf16(a1, b1, oacc[ot].v);
      }
    }
  }

  // final 1/l normalization, store bf16 into [row = b*Ls + p][col = h*64 + d]
  __syncthreads();
  if (tid < 64) rowScale[tid] = 1.0f / l_i;
  __syncthreads();
  {
    int mbase = 16 * wr + ((lane & 16) >> 1);
    int nn = lane & 15;
#pragma unroll
    for (int ot = 0; ot < 2; ++ot) {
      int gcol = h * 64 + 32 * wc + 16 * ot + nn;
#pragma unroll
      for (int v = 0; v < 8; ++v) {
        float val = oacc[ot].f[v] * rowScale[mbase + v];
        size_t row = (size_t)b * Ls + q0 + mbase + v;
        Obuf[row * (size_t)Dm + gcol] = f2bf_bits(val);
      }
    }
  }
}

// ---------------------------------------------------------------- launch
extern "C" void kernel_launch(void* const* d_in, const int* in_sizes, int n_in,
                              void* d_out, int out_size, void* d_ws, size_t ws_size,
                              hipStream_t stream) {
  const float* K_in      = (const float*)d_in[0];
  const float* V_in      = (const float*)d_in[1];
  const float* Q_in      = (const float*)d_in[2];
  const float* mask_attn = (const float*)d_in[3];
  const float* mask_pad  = (const float*)d_in[4];
  const float* Wk = (const float*)d_in[5];  const float* bk = (const float*)d_in[6];
  const float* Wv = (const float*)d_in[7];  const float* bv = (const float*)d_in[8];
  const float* Wq = (const float*)d_in[9];  const float* bq = (const float*)d_in[10];
  const float* Wo = (const float*)d_in[11]; const float* bo = (const float*)d_in[12];

  const int Mrows = Bb * Ls;                       // 4096
  const size_t xN = (size_t)Mrows * Dm;            // 4.19M elems
  const size_t wN = (size_t)Dm * Dm;               // 1.05M elems

  char* ws = (char*)d_ws;
  size_t off = 0;
  auto take = [&](size_t bytes) -> char* {
    char* p = ws + off; off += (bytes + 255) & ~(size_t)255; return p;
  };
  unsigned short* Xk  = (unsigned short*)take(xN * 2);
  unsigned short* Xv  = (unsigned short*)take(xN * 2);
  unsigned short* Xq  = (unsigned short*)take(xN * 2);
  unsigned short* Wkt = (unsigned short*)take(wN * 2);   // [N][K] transposed bf16
  unsigned short* Wvt = (unsigned short*)take(wN * 2);
  unsigned short* Wqt = (unsigned short*)take(wN * 2);
  unsigned short* Wot = (unsigned short*)take(wN * 2);
  unsigned short* Kp  = (unsigned short*)take(xN * 2);   // [b,h,l,dk]
  unsigned short* Vp  = (unsigned short*)take(xN * 2);   // [b,h,dk,l]
  unsigned short* Qp  = (unsigned short*)take(xN * 2);   // [b,h,lp,dk]
  unsigned short* Ob  = (unsigned short*)take(xN * 2);   // [b*lp, nh*dk]

  // 1) activations fp32 -> bf16; weights fp32 -> bf16 transposed [N][K]
  cvt_f32_bf16_kernel<<<dim3((unsigned)(xN / 4 / 256)), dim3(256), 0, stream>>>(K_in, Xk, (int)xN);
  cvt_f32_bf16_kernel<<<dim3((unsigned)(xN / 4 / 256)), dim3(256), 0, stream>>>(V_in, Xv, (int)xN);
  cvt_f32_bf16_kernel<<<dim3((unsigned)(xN / 4 / 256)), dim3(256), 0, stream>>>(Q_in, Xq, (int)xN);
  dim3 tgrid(Dm / 32, Dm / 32);
  cvt_transpose_bf16_kernel<<<tgrid, dim3(256), 0, stream>>>(Wk, Wkt, Dm, Dm);
  cvt_transpose_bf16_kernel<<<tgrid, dim3(256), 0, stream>>>(Wv, Wvt, Dm, Dm);
  cvt_transpose_bf16_kernel<<<tgrid, dim3(256), 0, stream>>>(Wq, Wqt, Dm, Dm);
  cvt_transpose_bf16_kernel<<<tgrid, dim3(256), 0, stream>>>(Wo, Wot, Dm, Dm);

  // 2) projections (WMMA GEMMs), Q pre-scaled by 1/sqrt(DK)=0.125
  dim3 ggrid(Dm / 128, Mrows / 128);
  gemm_bf16_wmma_kernel<<<ggrid, dim3(256), 0, stream>>>(Xk, Wkt, bk, Kp, Mrows, Dm, Dm, 1.0f,   0);
  gemm_bf16_wmma_kernel<<<ggrid, dim3(256), 0, stream>>>(Xv, Wvt, bv, Vp, Mrows, Dm, Dm, 1.0f,   1);
  gemm_bf16_wmma_kernel<<<ggrid, dim3(256), 0, stream>>>(Xq, Wqt, bq, Qp, Mrows, Dm, Dm, 0.125f, 0);

  // 3) flash attention per (q-tile, head, batch)
  attn_flash_kernel<<<dim3(Ls / 64, NH, Bb), dim3(256), 0, stream>>>(
      Qp, Kp, Vp, mask_attn, mask_pad, Ob);

  // 4) output projection -> f32 d_out
  gemm_bf16_wmma_kernel<<<ggrid, dim3(256), 0, stream>>>(Ob, Wot, bo, (float*)d_out,
                                                         Mrows, Dm, Dm, 1.0f, 2);
}
